// Bert_Traj_Model_54185307406799
// MI455X (gfx1250) — matrix operand order repaired and checked by reading, not compiled, not run on hardware
//
#include <hip/hip_runtime.h>
#include <math.h>

#define D_MODEL 768
#define N_HEADS 12
#define HEAD_DIM 64
#define NLAYERS 12
#define D_FF    3072
#define BATCH   32
#define SEQ     512
#define NTOK    (BATCH * SEQ)   // 16384
#define EPS     1e-6f

// GEMM tiling
#define BM 256
#define BN 128
#define BK 64

typedef __bf16 bf16_t;
typedef __attribute__((ext_vector_type(16))) __bf16 bf16x16;
typedef __attribute__((ext_vector_type(8)))  float  f32x8;

// ---------------------------------------------------------------- WMMA helper
__device__ __forceinline__ f32x8 wmma_bf16(bf16x16 a, bf16x16 b, f32x8 c) {
  // D = A(16x32 bf16) * B(32x16 bf16) + C(16x16 f32)
  return __builtin_amdgcn_wmma_f32_16x16x32_bf16(
      /*neg_a=*/false, a, /*neg_b=*/false, b,
      /*c_mod=*/(short)0, c, /*reuse_a=*/false, /*reuse_b=*/false);
}

// ---------------------------------------------------------------- async copy
// 16B global -> LDS DMA, per active lane; tracked with ASYNCcnt.
__device__ __forceinline__ void async_copy_b128(const bf16_t* g, bf16_t* l) {
  // Low 32 bits of the generic shared-aperture address are the LDS byte offset.
  const unsigned lds_addr = (unsigned)(size_t)l;
  asm volatile("global_load_async_to_lds_b128 %0, %1, off"
               :
               : "v"(lds_addr), "v"(g)
               : "memory");
}
__device__ __forceinline__ void wait_async0() {
  asm volatile("s_wait_asynccnt 0" ::: "memory");
}

// A-operand fragment from LDS, matrix stored row-major [m][k], row stride rs
// (in bf16 elems). ISA layout: lane L (row = L&15), VGPR0-3 hold K = 8*half + 0..7,
// VGPR4-7 hold K = 16 + 8*half + 0..7  -> two contiguous 16B runs.
__device__ __forceinline__ bf16x16 load_a_frag(const bf16_t* base, int rs, int lane) {
  const int lr = lane & 15, half = lane >> 4;
  const bf16_t* p = base + lr * rs + half * 8;
  union { bf16x16 v; uint4 q[2]; } u;
  u.q[0] = *reinterpret_cast<const uint4*>(p);
  u.q[1] = *reinterpret_cast<const uint4*>(p + 16);
  return u.v;
}

// B-operand fragment from LDS, matrix stored TRANSPOSED [n][k], row stride rs.
// ISA layout: lane L (col = L&15), VGPR0-3 hold K = 8*half + 0..7,
// VGPR4-7 hold K = 16 + 8*half + 0..7  -> two contiguous 16B runs.
__device__ __forceinline__ bf16x16 load_b_frag(const bf16_t* base, int rs, int lane) {
  const int lr = lane & 15, half = lane >> 4;
  const bf16_t* p = base + lr * rs + half * 8;
  union { bf16x16 v; uint4 q[2]; } u;
  u.q[0] = *reinterpret_cast<const uint4*>(p);
  u.q[1] = *reinterpret_cast<const uint4*>(p + 16);
  return u.v;
}

__device__ __forceinline__ f32x8 fzero8() {
  f32x8 z = {0.f, 0.f, 0.f, 0.f, 0.f, 0.f, 0.f, 0.f};
  return z;
}

// ---------------------------------------------------------------- block reduce
__device__ __forceinline__ float block_sum_256(float v, float* red, int tid) {
  red[tid] = v;
  __syncthreads();
  for (int s = 128; s > 0; s >>= 1) {
    if (tid < s) red[tid] += red[tid + s];
    __syncthreads();
  }
  float r = red[0];
  __syncthreads();
  return r;
}

// ---------------------------------------------------------------- weight cvt
// W [K][N] f32  ->  WT [N][K] bf16 (transposed so GEMM B-tiles are row chunks)
__global__ __launch_bounds__(256)
void wcvt_kernel(const float* __restrict__ W, bf16_t* __restrict__ WT,
                 int K, int N) {
  __shared__ bf16_t tile[64][65];
  const int n0 = blockIdx.x * 64, k0 = blockIdx.y * 64;
  const int tid = threadIdx.x;
  for (int e = tid; e < 64 * 64; e += 256) {
    const int kk = e >> 6, nn = e & 63;
    tile[kk][nn] = (bf16_t)W[(size_t)(k0 + kk) * N + n0 + nn];
  }
  __syncthreads();
  for (int e = tid; e < 64 * 64; e += 256) {
    const int nn = e >> 6, kk = e & 63;
    WT[(size_t)(n0 + nn) * K + k0 + kk] = tile[kk][nn];
  }
}

// ---------------------------------------------------------------- embed + LN
__global__ __launch_bounds__(256)
void embed_ln_kernel(const int* __restrict__ x, const int* __restrict__ t,
                     const float* __restrict__ tokE, const float* __restrict__ timE,
                     const float* __restrict__ g, const float* __restrict__ b,
                     float* __restrict__ h) {
  __shared__ float red[256];
  const int row = blockIdx.x;            // 0..NTOK-1
  const int tid = threadIdx.x;
  const int srow = row & (SEQ - 1);      // position within sequence
  const int xi = x[row], ti = t[row];
  float e[3];
  float lsum = 0.f;
#pragma unroll
  for (int j = 0; j < 3; ++j) {
    const int c = tid + j * 256;
    const int j2 = c >> 1;
    // 10000^(-(2*j2)/768) = exp(-(2*j2) * ln(10000)/768)
    const float freq = __expf(-(float)(2 * j2) * (9.210340371976184f / 768.0f));
    const float ang = (float)srow * freq;
    const float pe = (c & 1) ? __cosf(ang) : __sinf(ang);
    e[j] = tokE[(size_t)xi * D_MODEL + c] + timE[(size_t)ti * D_MODEL + c] + pe;
    lsum += e[j];
  }
  const float mu = block_sum_256(lsum, red, tid) * (1.f / 768.f);
  float lv = 0.f;
#pragma unroll
  for (int j = 0; j < 3; ++j) { float d = e[j] - mu; lv += d * d; }
  const float var = block_sum_256(lv, red, tid) * (1.f / 768.f);
  const float inv = rsqrtf(var + EPS);
#pragma unroll
  for (int j = 0; j < 3; ++j) {
    const int c = tid + j * 256;
    h[(size_t)row * D_MODEL + c] = g[c] * (e[j] - mu) * inv + b[c];
  }
}

// ---------------------------------------------------------------- LayerNorm -> bf16
__global__ __launch_bounds__(256)
void ln_bf16_kernel(const float* __restrict__ in, const float* __restrict__ g,
                    const float* __restrict__ b, bf16_t* __restrict__ out) {
  __shared__ float red[256];
  const int row = blockIdx.x;
  const int tid = threadIdx.x;
  float e[3];
  float lsum = 0.f;
#pragma unroll
  for (int j = 0; j < 3; ++j) {
    e[j] = in[(size_t)row * D_MODEL + tid + j * 256];
    lsum += e[j];
  }
  const float mu = block_sum_256(lsum, red, tid) * (1.f / 768.f);
  float lv = 0.f;
#pragma unroll
  for (int j = 0; j < 3; ++j) { float d = e[j] - mu; lv += d * d; }
  const float var = block_sum_256(lv, red, tid) * (1.f / 768.f);
  const float inv = rsqrtf(var + EPS);
#pragma unroll
  for (int j = 0; j < 3; ++j) {
    const int c = tid + j * 256;
    out[(size_t)row * D_MODEL + c] = (bf16_t)(g[c] * (e[j] - mu) * inv + b[c]);
  }
}

// ---------------------------------------------------------------- WMMA GEMM
// C[M,N] = act( A[M,K](bf16) * WT[N,K]^T(bf16) + bias[N] ) (+ res, f32)
// Block: 256 threads = 8 waves; tile 256(M) x 128(N); K-step 64.
// Waves in 4(M) x 2(N) grid; each wave owns a 64x64 macro-tile = 4x4 WMMA subtiles.
// Double-buffered LDS tiles filled with GLOBAL_LOAD_ASYNC_TO_LDS_B128: the DMA
// for tile k+1 overlaps the 32 WMMAs of tile k; one s_wait_asynccnt + barrier
// per K-step.
template <bool RELU, bool RES, bool OUTBF>
__global__ __launch_bounds__(256)
void gemm_kernel(const bf16_t* __restrict__ A, const bf16_t* __restrict__ WT,
                 const float* __restrict__ bias, const float* __restrict__ res,
                 void* __restrict__ Cv, int M, int N, int K) {
  __shared__ __align__(16) bf16_t s_a[2][BM][BK];    // 2 x 32 KB
  __shared__ __align__(16) bf16_t s_bt[2][BN][BK];   // 2 x 16 KB
  const int tid  = threadIdx.x;
  const int lane = tid & 31;
  const int wave = tid >> 5;
  const int wm = wave >> 1;   // 0..3 -> row offset wm*64
  const int wn = wave & 1;    // 0..1 -> col offset wn*64
  const int m0 = blockIdx.y * BM;
  const int n0 = blockIdx.x * BN;

  auto stage = [&](int buf, int kt) {
    for (int e = tid; e < BM * BK / 8; e += 256) {      // 2048 x 16B
      const int m = e >> 3, k8 = (e & 7) * 8;
      async_copy_b128(&A[(size_t)(m0 + m) * K + kt + k8], &s_a[buf][m][k8]);
    }
    for (int e = tid; e < BN * BK / 8; e += 256) {      // 1024 x 16B
      const int n = e >> 3, k8 = (e & 7) * 8;
      async_copy_b128(&WT[(size_t)(n0 + n) * K + kt + k8], &s_bt[buf][n][k8]);
    }
  };

  f32x8 acc[4][4];
#pragma unroll
  for (int i = 0; i < 4; ++i)
#pragma unroll
    for (int j = 0; j < 4; ++j) acc[i][j] = fzero8();

  // prologue: DMA tile 0
  stage(0, 0);
  wait_async0();
  __syncthreads();

  int cur = 0;
  for (int kt = 0; kt < K; kt += BK) {
    const int ktn = kt + BK;
    if (ktn < K) {
      // overlap: DMA next tile into the other buffer while computing this one
      stage(cur ^ 1, ktn);
      // keep L2 one more tile ahead (global_prefetch_b8)
      const int ktp = kt + 2 * BK;
      if (ktp < K) {
        __builtin_prefetch(&A[(size_t)(m0 + tid) * K + ktp], 0, 1);
        if (tid < BN) __builtin_prefetch(&WT[(size_t)(n0 + tid) * K + ktp], 0, 1);
      }
    }

#pragma unroll
    for (int kc = 0; kc < BK; kc += 32) {
      bf16x16 af[4], bf[4];
#pragma unroll
      for (int mi = 0; mi < 4; ++mi)
        af[mi] = load_a_frag(&s_a[cur][wm * 64 + mi * 16][kc], BK, lane);
#pragma unroll
      for (int ni = 0; ni < 4; ++ni)
        bf[ni] = load_b_frag(&s_bt[cur][wn * 64 + ni * 16][kc], BK, lane);
#pragma unroll
      for (int mi = 0; mi < 4; ++mi)
#pragma unroll
        for (int ni = 0; ni < 4; ++ni)
          acc[mi][ni] = wmma_bf16(af[mi], bf[ni], acc[mi][ni]);
    }

    wait_async0();       // next tile's DMA (overlapped with WMMAs) has landed
    __syncthreads();     // all waves done reading cur + all DMA visible
    cur ^= 1;
  }

  const int lr = lane & 15, half = lane >> 4;
#pragma unroll
  for (int mi = 0; mi < 4; ++mi) {
#pragma unroll
    for (int ni = 0; ni < 4; ++ni) {
      const int col = n0 + wn * 64 + ni * 16 + lr;
      const float bv = bias[col];
#pragma unroll
      for (int r = 0; r < 8; ++r) {
        const int row = m0 + wm * 64 + mi * 16 + r + 8 * half;
        float vv = acc[mi][ni][r] + bv;
        if (RELU) vv = fmaxf(vv, 0.f);
        const size_t idx = (size_t)row * N + col;
        if (RES) vv += res[idx];
        if (OUTBF) ((bf16_t*)Cv)[idx] = (bf16_t)vv;
        else       ((float*)Cv)[idx]  = vv;
      }
    }
  }
}

// ---------------------------------------------------------------- attention
// One workgroup: (batch b, head hh, 16 query rows). Block = 128 threads (4 waves).
// q/k/v are bf16. Scores 16x512 via WMMA with double-buffered async K staging,
// masked+scaled, row softmax in LDS, P(bf16) x V via WMMA (V transposed into LDS).
__global__ __launch_bounds__(128)
void attn_kernel(const bf16_t* __restrict__ q, const bf16_t* __restrict__ kmat,
                 const bf16_t* __restrict__ vmat, const int* __restrict__ x,
                 const int* __restrict__ lenp, bf16_t* __restrict__ o) {
  __shared__ __align__(16) bf16_t s_q[16][64];       // Q tile  [m][d]
  __shared__ __align__(16) bf16_t s_k[2][64][64];    // K chunks [key][d], double buffer
  __shared__ __align__(16) bf16_t s_vt[64][64];      // V chunk transposed [d][key]
  __shared__ __align__(16) float  s_s[16][512];      // scores f32
  __shared__ __align__(16) bf16_t s_p[16][512];      // softmax probs bf16

  const int tid = threadIdx.x, lane = tid & 31, w = tid >> 5;
  const int b = blockIdx.z, hh = blockIdx.y, q0 = blockIdx.x * 16;
  const int lt = lenp[0];
  const int lr = lane & 15, half = lane >> 4;

  auto stage_k = [&](int buf, int kk) {
    for (int e = tid; e < 64 * 8; e += 128) {        // 512 x 16B ops
      const int key = e >> 3, d8 = (e & 7) * 8;
      async_copy_b128(
          &kmat[(size_t)(b * SEQ + kk + key) * D_MODEL + hh * HEAD_DIM + d8],
          &s_k[buf][key][d8]);
    }
  };

  // prologue: async DMA Q tile (128 x 16B, one per thread) + K chunk 0
  {
    const int m = tid >> 3, d8 = (tid & 7) * 8;
    async_copy_b128(&q[(size_t)(b * SEQ + q0 + m) * D_MODEL + hh * HEAD_DIM + d8],
                    &s_q[m][d8]);
  }
  stage_k(0, 0);
  wait_async0();
  __syncthreads();

  // ---- phase 1: scores = (Q K^T) * 0.125, masked ----
  int cur = 0;
  for (int kk = 0; kk < SEQ; kk += 64) {
    if (kk + 64 < SEQ) stage_k(cur ^ 1, kk + 64);    // overlap next K chunk

    f32x8 acc = fzero8();
#pragma unroll
    for (int kc = 0; kc < 64; kc += 32) {
      bf16x16 a  = load_a_frag(&s_q[0][kc], 64, lane);
      bf16x16 bb = load_b_frag(&s_k[cur][w * 16][kc], 64, lane);
      acc = wmma_bf16(a, bb, acc);
    }

    const int keyabs = kk + w * 16 + lr;
    const bool pad = x[b * SEQ + keyabs] > 0;
    const bool pre = keyabs < lt;
#pragma unroll
    for (int r = 0; r < 8; ++r) {
      const int qrow = q0 + r + 8 * half;
      const bool vis = pad && (pre || keyabs <= qrow);
      s_s[r + 8 * half][kk + w * 16 + lr] = vis ? acc[r] * 0.125f : -1e9f;
    }

    wait_async0();
    __syncthreads();
    cur ^= 1;
  }

  // ---- softmax per row (16 rows, one thread each) ----
  if (tid < 16) {
    float mx = -3.0e38f;
    for (int c = 0; c < SEQ; ++c) mx = fmaxf(mx, s_s[tid][c]);
    float sum = 0.f;
    for (int c = 0; c < SEQ; ++c) {
      const float e = __expf(s_s[tid][c] - mx);
      s_s[tid][c] = e;
      sum += e;
    }
    const float inv = 1.f / sum;
    for (int c = 0; c < SEQ; ++c) s_p[tid][c] = (bf16_t)(s_s[tid][c] * inv);
  }
  __syncthreads();

  // ---- phase 2: O = P * V ----
  f32x8 oacc = fzero8();
  for (int kk = 0; kk < SEQ; kk += 64) {
    // V chunk: read 8 bf16 (16B) along d, scatter-transpose into [d][key]
    for (int e = tid; e < 64 * 8; e += 128) {
      const int key = e >> 3, d8 = (e & 7) * 8;
      union { uint4 u; bf16_t hv[8]; } pk;
      pk.u = *reinterpret_cast<const uint4*>(
          &vmat[(size_t)(b * SEQ + kk + key) * D_MODEL + hh * HEAD_DIM + d8]);
#pragma unroll
      for (int j = 0; j < 8; ++j) s_vt[d8 + j][key] = pk.hv[j];
    }
    __syncthreads();
#pragma unroll
    for (int kc = 0; kc < 64; kc += 32) {
      bf16x16 a  = load_a_frag(&s_p[0][kk + kc], 512, lane);
      bf16x16 bb = load_b_frag(&s_vt[w * 16][kc], 64, lane);
      oacc = wmma_bf16(a, bb, oacc);
    }
    __syncthreads();
  }

  const int col = hh * HEAD_DIM + w * 16 + lr;
#pragma unroll
  for (int r = 0; r < 8; ++r) {
    const int row = q0 + r + 8 * half;
    o[(size_t)(b * SEQ + row) * D_MODEL + col] = (bf16_t)oacc[r];
  }
}

// ---------------------------------------------------------------- launcher
extern "C" void kernel_launch(void* const* d_in, const int* in_sizes, int n_in,
                              void* d_out, int out_size, void* d_ws, size_t ws_size,
                              hipStream_t stream) {
  const int*   x    = (const int*)d_in[0];
  const int*   tim  = (const int*)d_in[1];
  const int*   lt   = (const int*)d_in[2];
  const float* tokE = (const float*)d_in[3];
  const float* timE = (const float*)d_in[4];
  const float* eg   = (const float*)d_in[5];
  const float* eb   = (const float*)d_in[6];
  const float* Wq   = (const float*)d_in[7];
  const float* bq   = (const float*)d_in[8];
  const float* Wk   = (const float*)d_in[9];
  const float* bk   = (const float*)d_in[10];
  const float* Wv   = (const float*)d_in[11];
  const float* bv   = (const float*)d_in[12];
  const float* Wo   = (const float*)d_in[13];
  const float* bo   = (const float*)d_in[14];
  const float* l1g  = (const float*)d_in[15];
  const float* l1b  = (const float*)d_in[16];
  const float* W1   = (const float*)d_in[17];
  const float* b1   = (const float*)d_in[18];
  const float* W2   = (const float*)d_in[19];
  const float* b2   = (const float*)d_in[20];
  const float* l2g  = (const float*)d_in[21];
  const float* l2b  = (const float*)d_in[22];

  float* h = (float*)d_out;           // h (f32) lives in d_out across the forward
  const size_t MD = (size_t)NTOK * D_MODEL;
  const size_t MF = (size_t)NTOK * D_FF;
  const size_t DD = (size_t)D_MODEL * D_MODEL;
  const size_t DF = (size_t)D_MODEL * D_FF;

  char* p = (char*)d_ws;
  bf16_t* hn  = (bf16_t*)p; p += MD * 2;
  bf16_t* qb  = (bf16_t*)p; p += MD * 2;
  bf16_t* kb  = (bf16_t*)p; p += MD * 2;
  bf16_t* vb  = (bf16_t*)p; p += MD * 2;
  bf16_t* ob  = (bf16_t*)p; p += MD * 2;
  bf16_t* f1  = (bf16_t*)p; p += MF * 2;
  bf16_t* WTq = (bf16_t*)p; p += NLAYERS * DD * 2;
  bf16_t* WTk = (bf16_t*)p; p += NLAYERS * DD * 2;
  bf16_t* WTv = (bf16_t*)p; p += NLAYERS * DD * 2;
  bf16_t* WTo = (bf16_t*)p; p += NLAYERS * DD * 2;
  bf16_t* WT1 = (bf16_t*)p; p += NLAYERS * DF * 2;
  bf16_t* WT2 = (bf16_t*)p; p += NLAYERS * DF * 2;

  const dim3 blk256(256), blk128(128);
  const dim3 g768(D_MODEL / BN, NTOK / BM);
  const dim3 g3072(D_FF / BN, NTOK / BM);
  const dim3 gattn(SEQ / 16, N_HEADS, BATCH);
  const dim3 gc_dd(D_MODEL / 64, D_MODEL / 64);   // W [768][768]
  const dim3 gc_1(D_FF / 64, D_MODEL / 64);       // W1 [768][3072]
  const dim3 gc_2(D_MODEL / 64, D_FF / 64);       // W2 [3072][768]

  // one-time (per launch) weight convert + transpose to bf16 [N][K]
  for (int i = 0; i < NLAYERS; ++i) {
    wcvt_kernel<<<gc_dd, blk256, 0, stream>>>(Wq + i * DD, WTq + i * DD, D_MODEL, D_MODEL);
    wcvt_kernel<<<gc_dd, blk256, 0, stream>>>(Wk + i * DD, WTk + i * DD, D_MODEL, D_MODEL);
    wcvt_kernel<<<gc_dd, blk256, 0, stream>>>(Wv + i * DD, WTv + i * DD, D_MODEL, D_MODEL);
    wcvt_kernel<<<gc_dd, blk256, 0, stream>>>(Wo + i * DD, WTo + i * DD, D_MODEL, D_MODEL);
    wcvt_kernel<<<gc_1,  blk256, 0, stream>>>(W1 + i * DF, WT1 + i * DF, D_MODEL, D_FF);
    wcvt_kernel<<<gc_2,  blk256, 0, stream>>>(W2 + i * DF, WT2 + i * DF, D_FF, D_MODEL);
  }

  embed_ln_kernel<<<NTOK, blk256, 0, stream>>>(x, tim, tokE, timE, eg, eb, h);

  for (int i = 0; i < NLAYERS; ++i) {
    ln_bf16_kernel<<<NTOK, blk256, 0, stream>>>(h, l1g + i * D_MODEL, l1b + i * D_MODEL, hn);

    gemm_kernel<false, false, true><<<g768, blk256, 0, stream>>>(
        hn, WTq + i * DD, bq + i * D_MODEL, nullptr, qb, NTOK, D_MODEL, D_MODEL);
    gemm_kernel<false, false, true><<<g768, blk256, 0, stream>>>(
        hn, WTk + i * DD, bk + i * D_MODEL, nullptr, kb, NTOK, D_MODEL, D_MODEL);
    gemm_kernel<false, false, true><<<g768, blk256, 0, stream>>>(
        hn, WTv + i * DD, bv + i * D_MODEL, nullptr, vb, NTOK, D_MODEL, D_MODEL);

    attn_kernel<<<gattn, blk128, 0, stream>>>(qb, kb, vb, x, lt, ob);

    gemm_kernel<false, true, false><<<g768, blk256, 0, stream>>>(
        ob, WTo + i * DD, bo + i * D_MODEL, h, h, NTOK, D_MODEL, D_MODEL);

    ln_bf16_kernel<<<NTOK, blk256, 0, stream>>>(h, l2g + i * D_MODEL, l2b + i * D_MODEL, hn);

    gemm_kernel<true, false, true><<<g3072, blk256, 0, stream>>>(
        hn, WT1 + i * DF, b1 + i * D_FF, nullptr, f1, NTOK, D_FF, D_MODEL);

    gemm_kernel<false, true, false><<<g768, blk256, 0, stream>>>(
        f1, WT2 + i * DF, b2 + i * D_MODEL, h, h, NTOK, D_MODEL, D_FF);
  }
}
